// JambaMambaMixer_70248485094062
// MI455X (gfx1250) — compile-verified, matrix operand order checked by reading, not run on hardware
//
#include <hip/hip_runtime.h>
#include <hip/hip_bf16.h>

// ---------------- shapes ----------------
#define S_LEN 1024
#define H_DIM 2048
#define I_DIM 4096
#define N_ST  16
#define R_DIM 128
#define J_DIM 8192   // 2*I
#define G_DIM 160    // R + 2N
#define CH    8      // scan chunks
#define CS    (S_LEN / CH)   // 128 steps per chunk

// ---------------- WMMA types ----------------
typedef __attribute__((ext_vector_type(16))) __bf16    v16bf;
typedef __attribute__((ext_vector_type(8)))  float     v8f;
typedef __attribute__((ext_vector_type(4)))  unsigned  uint4v;
typedef __attribute__((ext_vector_type(4)))  float     float4v;

union FragCast { v16bf bf; uint4v q[2]; };
union AccCast  { v8f v; float4v q[2]; float f[8]; };

__device__ __forceinline__ unsigned short f2bf(float x){
  unsigned u = __builtin_bit_cast(unsigned, x);
  u += 0x7FFFu + ((u >> 16) & 1u);       // round-to-nearest-even
  return (unsigned short)(u >> 16);
}

// A fragment (16x32 bf16, M x K) and B fragment (32x16 bf16, K x N) share the
// same per-lane load pattern: lane&15 selects the row/col, lane>>4 selects the
// K half; 16 contiguous bf16 = two b128 loads.
__device__ __forceinline__ v16bf load_frag(const unsigned short* __restrict__ base,
                                           int ld, int rc0, int k0, int lane){
  const unsigned short* p = base + (size_t)(rc0 + (lane & 15)) * ld
                                 + (size_t)(k0 + ((lane >> 4) << 4));
  FragCast f;
  f.q[0] = *(const uint4v*)p;
  f.q[1] = *(const uint4v*)(p + 8);
  return f.bf;
}

__device__ __forceinline__ v8f wmma_bf16(v16bf a, v16bf b, v8f c){
  return __builtin_amdgcn_wmma_f32_16x16x32_bf16(false, a, false, b, (short)0, c,
                                                 false, false);
}

// ---------------- fp32 -> bf16 convert ----------------
__global__ void k_cvt(const float* __restrict__ in, unsigned short* __restrict__ out, int n){
  int i = blockIdx.x * 256 + threadIdx.x;
  if (i < n) out[i] = f2bf(in[i]);
}

// ============================================================================
// Big GEMM skeleton: 32(M) x 64(N) per wave, software-pipelined K loop.
// Computes D[m0..m0+31, s0..s0+63] = A[MxK] * B[NxK]^T, all bf16 K-contiguous.
// Epilogue handled by caller-specific kernels below.
// ============================================================================
#define GEMM_PIPE(Aptr, Bptr, LD, KDIM)                                        \
  v8f acc[8] = {};                                                             \
  v16bf a0 = load_frag(Aptr, LD, m0,      0, lane);                            \
  v16bf a1 = load_frag(Aptr, LD, m0 + 16, 0, lane);                            \
  v16bf b0 = load_frag(Bptr, LD, s0,      0, lane);                            \
  v16bf b1 = load_frag(Bptr, LD, s0 + 16, 0, lane);                            \
  v16bf b2 = load_frag(Bptr, LD, s0 + 32, 0, lane);                            \
  v16bf b3 = load_frag(Bptr, LD, s0 + 48, 0, lane);                            \
  for (int k = 32; k < (KDIM); k += 32){                                       \
    __builtin_prefetch(Aptr + (size_t)(m0 + (lane & 15)) * (LD) + k + 96, 0, 3);\
    v16bf na0 = load_frag(Aptr, LD, m0,      k, lane);                         \
    v16bf na1 = load_frag(Aptr, LD, m0 + 16, k, lane);                         \
    v16bf nb0 = load_frag(Bptr, LD, s0,      k, lane);                         \
    v16bf nb1 = load_frag(Bptr, LD, s0 + 16, k, lane);                         \
    v16bf nb2 = load_frag(Bptr, LD, s0 + 32, k, lane);                         \
    v16bf nb3 = load_frag(Bptr, LD, s0 + 48, k, lane);                         \
    acc[0] = wmma_bf16(a0, b0, acc[0]);                                        \
    acc[1] = wmma_bf16(a0, b1, acc[1]);                                        \
    acc[2] = wmma_bf16(a0, b2, acc[2]);                                        \
    acc[3] = wmma_bf16(a0, b3, acc[3]);                                        \
    acc[4] = wmma_bf16(a1, b0, acc[4]);                                        \
    acc[5] = wmma_bf16(a1, b1, acc[5]);                                        \
    acc[6] = wmma_bf16(a1, b2, acc[6]);                                        \
    acc[7] = wmma_bf16(a1, b3, acc[7]);                                        \
    a0 = na0; a1 = na1; b0 = nb0; b1 = nb1; b2 = nb2; b3 = nb3;                \
  }                                                                            \
  acc[0] = wmma_bf16(a0, b0, acc[0]);                                          \
  acc[1] = wmma_bf16(a0, b1, acc[1]);                                          \
  acc[2] = wmma_bf16(a0, b2, acc[2]);                                          \
  acc[3] = wmma_bf16(a0, b3, acc[3]);                                          \
  acc[4] = wmma_bf16(a1, b0, acc[4]);                                          \
  acc[5] = wmma_bf16(a1, b1, acc[5]);                                          \
  acc[6] = wmma_bf16(a1, b2, acc[6]);                                          \
  acc[7] = wmma_bf16(a1, b3, acc[7]);

// ---------------- in_proj: proj[j,s] = sum_h W[j,h] * X[s,h] ----------------
// j < I  -> hs_pre_t[s*I + j]   (s-major for the conv)
// j >= I -> gate_t [s*I + j-I]  (s-major for the scan epilogue)
__global__ void __launch_bounds__(256)
k_inproj(const unsigned short* __restrict__ Wb, const unsigned short* __restrict__ Xb,
         float* __restrict__ hs_pre_t, float* __restrict__ gate_t){
  const int lane = threadIdx.x & 31, wave = threadIdx.x >> 5;
  const int m0 = (blockIdx.x * 8 + wave) * 32;
  const int s0 = blockIdx.y * 64;
  GEMM_PIPE(Wb, Xb, H_DIM, H_DIM)
  const int n = lane & 15, mb = (lane >> 4) * 8;
  #pragma unroll
  for (int half = 0; half < 2; ++half){
    const int j = m0 + 16 * half + mb;
    #pragma unroll
    for (int t = 0; t < 4; ++t){
      const int s = s0 + 16 * t + n;
      AccCast a; a.v = acc[half * 4 + t];
      float* dst = (j < I_DIM) ? (hs_pre_t + (size_t)s * I_DIM + j)
                               : (gate_t   + (size_t)s * I_DIM + (j - I_DIM));
      *(float4v*)dst       = a.q[0];
      *(float4v*)(dst + 4) = a.q[1];
    }
  }
}

// ---------------- out_proj: out[s,h] = sum_i so[s,i] * Wo[h,i] --------------
__global__ void __launch_bounds__(256)
k_outproj(const unsigned short* __restrict__ Wob, const unsigned short* __restrict__ Sob,
          float* __restrict__ out){
  const int lane = threadIdx.x & 31, wave = threadIdx.x >> 5;
  const int m0 = (blockIdx.x * 8 + wave) * 32;
  const int s0 = blockIdx.y * 64;
  GEMM_PIPE(Wob, Sob, I_DIM, I_DIM)
  const int n = lane & 15, mb = (lane >> 4) * 8;
  #pragma unroll
  for (int half = 0; half < 2; ++half){
    const int h = m0 + 16 * half + mb;
    #pragma unroll
    for (int t = 0; t < 4; ++t){
      const int s = s0 + 16 * t + n;
      AccCast a; a.v = acc[half * 4 + t];
      float* dst = out + (size_t)s * H_DIM + h;
      *(float4v*)dst       = a.q[0];
      *(float4v*)(dst + 4) = a.q[1];
    }
  }
}

// ---------------- causal depthwise conv (K=4) + SiLU + mask ----------------
__global__ void __launch_bounds__(256)
k_conv(const float* __restrict__ hs_pre_t, const float* __restrict__ mask,
       const float* __restrict__ conv_w, const float* __restrict__ conv_b,
       float* __restrict__ hs_t, unsigned short* __restrict__ hs_tb){
  const int i = blockIdx.x * 256 + threadIdx.x;
  const float w0 = conv_w[i*4+0], w1 = conv_w[i*4+1], w2 = conv_w[i*4+2], w3 = conv_w[i*4+3];
  const float b = conv_b[i];
  float x0 = 0.f, x1 = 0.f, x2 = 0.f;             // s-3, s-2, s-1
  for (int s = 0; s < S_LEN; ++s){
    const float m  = mask[s];
    const float x3 = hs_pre_t[(size_t)s * I_DIM + i] * m;   // pre-conv mask
    const float y  = w3*x3 + w2*x2 + w1*x1 + w0*x0 + b;
    const float h  = (y / (1.f + __expf(-y))) * m;          // silu, post mask
    hs_t [(size_t)s * I_DIM + i] = h;
    hs_tb[(size_t)s * I_DIM + i] = f2bf(h);
    x0 = x1; x1 = x2; x2 = x3;
  }
}

// ---------------- x_proj: ssm[s,j] = sum_i hs[i,s] * Wx[j,i] (j<160) --------
__global__ void __launch_bounds__(320)
k_xproj(const unsigned short* __restrict__ Wxb, const unsigned short* __restrict__ Hb,
        float* __restrict__ ssm){
  const int lane = threadIdx.x & 31, wave = threadIdx.x >> 5;  // 10 waves
  const int j0 = wave * 16;
  const int s0 = blockIdx.y * 64;
  v8f acc[4] = {};
  for (int k = 0; k < I_DIM; k += 32){
    v16bf a  = load_frag(Wxb, I_DIM, j0, k, lane);
    v16bf b0 = load_frag(Hb, I_DIM, s0,      k, lane);
    v16bf b1 = load_frag(Hb, I_DIM, s0 + 16, k, lane);
    v16bf b2 = load_frag(Hb, I_DIM, s0 + 32, k, lane);
    v16bf b3 = load_frag(Hb, I_DIM, s0 + 48, k, lane);
    acc[0] = wmma_bf16(a, b0, acc[0]);
    acc[1] = wmma_bf16(a, b1, acc[1]);
    acc[2] = wmma_bf16(a, b2, acc[2]);
    acc[3] = wmma_bf16(a, b3, acc[3]);
  }
  const int n = lane & 15, mb = (lane >> 4) * 8;
  #pragma unroll
  for (int t = 0; t < 4; ++t){
    const int s = s0 + 16 * t + n;
    AccCast a; a.v = acc[t];
    float* dst = ssm + (size_t)s * G_DIM + j0 + mb;
    *(float4v*)dst       = a.q[0];
    *(float4v*)(dst + 4) = a.q[1];
  }
}

// ---------------- rmsnorm of dt(128) / B(16) / C(16) per position -----------
__global__ void __launch_bounds__(160)
k_rms(const float* __restrict__ ssm, const float* __restrict__ dt_ln,
      const float* __restrict__ b_ln, const float* __restrict__ c_ln,
      unsigned short* __restrict__ tsb, float* __restrict__ Bm, float* __restrict__ Cm){
  const int s = blockIdx.x, t = threadIdx.x;
  __shared__ float red[G_DIM];
  __shared__ float inv[3];
  const float x = ssm[(size_t)s * G_DIM + t];
  red[t] = x * x;
  __syncthreads();
  if (t == 0){ float a = 0.f; for (int k = 0;   k < 128; ++k) a += red[k]; inv[0] = rsqrtf(a * (1.f/128.f) + 1e-6f); }
  if (t == 1){ float a = 0.f; for (int k = 128; k < 144; ++k) a += red[k]; inv[1] = rsqrtf(a * (1.f/16.f)  + 1e-6f); }
  if (t == 2){ float a = 0.f; for (int k = 144; k < 160; ++k) a += red[k]; inv[2] = rsqrtf(a * (1.f/16.f)  + 1e-6f); }
  __syncthreads();
  if (t < 128)      tsb[(size_t)s * R_DIM + t]      = f2bf(x * inv[0] * dt_ln[t]);
  else if (t < 144) Bm [(size_t)s * N_ST + (t-128)] = x * inv[1] * b_ln[t-128];
  else              Cm [(size_t)s * N_ST + (t-144)] = x * inv[2] * c_ln[t-144];
}

// ---------------- dt_proj + bias + softplus -> dt_t[s*I + i] ----------------
__global__ void __launch_bounds__(256)
k_dtproj(const unsigned short* __restrict__ Wdb, const unsigned short* __restrict__ Tsb,
         const float* __restrict__ bias, float* __restrict__ dt_t){
  const int lane = threadIdx.x & 31, wave = threadIdx.x >> 5;
  const int i0 = (blockIdx.x * 8 + wave) * 16;
  const int s0 = blockIdx.y * 64;
  v8f acc[4] = {};
  #pragma unroll
  for (int k = 0; k < R_DIM; k += 32){
    v16bf a  = load_frag(Wdb, R_DIM, i0, k, lane);
    v16bf b0 = load_frag(Tsb, R_DIM, s0,      k, lane);
    v16bf b1 = load_frag(Tsb, R_DIM, s0 + 16, k, lane);
    v16bf b2 = load_frag(Tsb, R_DIM, s0 + 32, k, lane);
    v16bf b3 = load_frag(Tsb, R_DIM, s0 + 48, k, lane);
    acc[0] = wmma_bf16(a, b0, acc[0]);
    acc[1] = wmma_bf16(a, b1, acc[1]);
    acc[2] = wmma_bf16(a, b2, acc[2]);
    acc[3] = wmma_bf16(a, b3, acc[3]);
  }
  const int n = lane & 15, mb = (lane >> 4) * 8;
  float4v bv0 = *(const float4v*)(bias + i0 + mb);
  float4v bv1 = *(const float4v*)(bias + i0 + mb + 4);
  #pragma unroll
  for (int t = 0; t < 4; ++t){
    const int s = s0 + 16 * t + n;
    AccCast a; a.v = acc[t];
    #pragma unroll
    for (int e = 0; e < 8; ++e){
      float xv = a.f[e] + ((e < 4) ? bv0[e] : bv1[e-4]);
      a.f[e] = (xv > 20.f) ? xv : log1pf(__expf(xv));       // softplus
    }
    float* dst = dt_t + (size_t)s * I_DIM + i0 + mb;
    *(float4v*)dst       = a.q[0];
    *(float4v*)(dst + 4) = a.q[1];
  }
}

// ============================================================================
// Chunked parallel scan (linear recurrence): S split into CH chunks of CS.
// Phase 1: per (channel, chunk) compute P = prod(dA), q = chunk-end state
//          starting from zero state.
// Phase 2: per (channel, n) sequentially combine chunk boundaries.
// Phase 3: per (channel, chunk) replay from the correct initial state, emit y.
// B/C for the chunk are staged in LDS (shared by the whole block).
// ============================================================================
__global__ void __launch_bounds__(256)
k_scan_part(const float* __restrict__ dt_t, const float* __restrict__ hs_t,
            const float* __restrict__ Bm, const float* __restrict__ A_log,
            float* __restrict__ Pst, float* __restrict__ Qst){
  const int i = blockIdx.x * 256 + threadIdx.x;
  const int c = blockIdx.y;
  __shared__ float Bsh[CS * N_ST];
  for (int t = threadIdx.x; t < CS * N_ST; t += 256)
    Bsh[t] = Bm[(size_t)c * CS * N_ST + t];
  __syncthreads();
  float A[N_ST], P[N_ST], q[N_ST];
  #pragma unroll
  for (int nn = 0; nn < N_ST; ++nn){
    A[nn] = -__expf(A_log[(size_t)i * N_ST + nn]); P[nn] = 1.f; q[nn] = 0.f;
  }
  for (int sl = 0; sl < CS; ++sl){
    const int s = c * CS + sl;
    const float dt = dt_t[(size_t)s * I_DIM + i];
    const float h  = hs_t[(size_t)s * I_DIM + i];
    const float dth = dt * h;
    #pragma unroll
    for (int nn = 0; nn < N_ST; ++nn){
      const float dA = __expf(A[nn] * dt);
      P[nn] *= dA;
      q[nn] = dA * q[nn] + dth * Bsh[sl * N_ST + nn];
    }
  }
  float* pd = Pst + ((size_t)i * CH + c) * N_ST;
  float* qd = Qst + ((size_t)i * CH + c) * N_ST;
  #pragma unroll
  for (int nn = 0; nn < N_ST; ++nn){ pd[nn] = P[nn]; qd[nn] = q[nn]; }
}

__global__ void __launch_bounds__(256)
k_scan_combine(const float* __restrict__ Pst, const float* __restrict__ Qst,
               float* __restrict__ S0){
  const int idx = blockIdx.x * 256 + threadIdx.x;   // i*N + n
  const int i = idx >> 4, nn = idx & 15;
  float st = 0.f;
  #pragma unroll
  for (int c = 0; c < CH; ++c){
    S0[((size_t)i * CH + c) * N_ST + nn] = st;
    st = Pst[((size_t)i * CH + c) * N_ST + nn] * st
       + Qst[((size_t)i * CH + c) * N_ST + nn];
  }
}

__global__ void __launch_bounds__(256)
k_scan_final(const float* __restrict__ dt_t, const float* __restrict__ hs_t,
             const float* __restrict__ Bm, const float* __restrict__ Cm,
             const float* __restrict__ A_log, const float* __restrict__ Dp,
             const float* __restrict__ gate_t, const float* __restrict__ S0,
             unsigned short* __restrict__ sob){
  const int i = blockIdx.x * 256 + threadIdx.x;
  const int c = blockIdx.y;
  __shared__ float Bsh[CS * N_ST];
  __shared__ float Csh[CS * N_ST];
  for (int t = threadIdx.x; t < CS * N_ST; t += 256){
    Bsh[t] = Bm[(size_t)c * CS * N_ST + t];
    Csh[t] = Cm[(size_t)c * CS * N_ST + t];
  }
  __syncthreads();
  float A[N_ST], st[N_ST];
  const float* s0p = S0 + ((size_t)i * CH + c) * N_ST;
  #pragma unroll
  for (int nn = 0; nn < N_ST; ++nn){
    A[nn] = -__expf(A_log[(size_t)i * N_ST + nn]); st[nn] = s0p[nn];
  }
  const float Dv = Dp[i];
  for (int sl = 0; sl < CS; ++sl){
    const int s = c * CS + sl;
    const float dt = dt_t[(size_t)s * I_DIM + i];
    const float h  = hs_t[(size_t)s * I_DIM + i];
    const float dth = dt * h;
    float y = 0.f;
    #pragma unroll
    for (int nn = 0; nn < N_ST; ++nn){
      const float dA = __expf(A[nn] * dt);
      st[nn] = dA * st[nn] + dth * Bsh[sl * N_ST + nn];
      y += st[nn] * Csh[sl * N_ST + nn];
    }
    y += Dv * h;
    const float g = gate_t[(size_t)s * I_DIM + i];
    const float so = y * (g / (1.f + __expf(-g)));           // * silu(gate)
    sob[(size_t)s * I_DIM + i] = f2bf(so);
  }
}

// ---------------- launcher ----------------
extern "C" void kernel_launch(void* const* d_in, const int* in_sizes, int n_in,
                              void* d_out, int out_size, void* d_ws, size_t ws_size,
                              hipStream_t stream){
  (void)in_sizes; (void)n_in; (void)out_size; (void)ws_size;
  const float* x      = (const float*)d_in[0];
  const float* mask   = (const float*)d_in[1];
  const float* w_in   = (const float*)d_in[2];
  const float* conv_w = (const float*)d_in[3];
  const float* conv_b = (const float*)d_in[4];
  const float* w_xp   = (const float*)d_in[5];
  const float* w_dt   = (const float*)d_in[6];
  const float* b_dt   = (const float*)d_in[7];
  const float* dt_ln  = (const float*)d_in[8];
  const float* b_ln   = (const float*)d_in[9];
  const float* c_ln   = (const float*)d_in[10];
  const float* A_log  = (const float*)d_in[11];
  const float* Dp     = (const float*)d_in[12];
  const float* w_out  = (const float*)d_in[13];
  float* out = (float*)d_out;

  char* p = (char*)d_ws;
  auto alloc = [&](size_t bytes) -> void* {
    void* r = (void*)p; p += (bytes + 255) & ~(size_t)255; return r;
  };
  unsigned short* xb    = (unsigned short*)alloc((size_t)S_LEN * H_DIM * 2);
  unsigned short* wib   = (unsigned short*)alloc((size_t)J_DIM * H_DIM * 2);
  unsigned short* xwb   = (unsigned short*)alloc((size_t)G_DIM * I_DIM * 2);
  unsigned short* dtwb  = (unsigned short*)alloc((size_t)I_DIM * R_DIM * 2);
  unsigned short* wob   = (unsigned short*)alloc((size_t)H_DIM * I_DIM * 2);
  float* hs_pre_t       = (float*)alloc((size_t)S_LEN * I_DIM * 4);
  float* gate_t         = (float*)alloc((size_t)S_LEN * I_DIM * 4);
  float* hs_t           = (float*)alloc((size_t)S_LEN * I_DIM * 4);
  unsigned short* hs_tb = (unsigned short*)alloc((size_t)S_LEN * I_DIM * 2);
  float* ssm            = (float*)alloc((size_t)S_LEN * G_DIM * 4);
  unsigned short* tsb   = (unsigned short*)alloc((size_t)S_LEN * R_DIM * 2);
  float* Bmat           = (float*)alloc((size_t)S_LEN * N_ST * 4);
  float* Cmat           = (float*)alloc((size_t)S_LEN * N_ST * 4);
  float* dt_t           = (float*)alloc((size_t)S_LEN * I_DIM * 4);
  unsigned short* sob   = (unsigned short*)alloc((size_t)S_LEN * I_DIM * 2);
  float* Pst            = (float*)alloc((size_t)I_DIM * CH * N_ST * 4);
  float* Qst            = (float*)alloc((size_t)I_DIM * CH * N_ST * 4);
  float* S0             = (float*)alloc((size_t)I_DIM * CH * N_ST * 4);

  auto cvt = [&](const float* in, unsigned short* o, int n){
    k_cvt<<<(n + 255) / 256, 256, 0, stream>>>(in, o, n);
  };
  cvt(x,     xb,   S_LEN * H_DIM);
  cvt(w_in,  wib,  J_DIM * H_DIM);
  cvt(w_xp,  xwb,  G_DIM * I_DIM);
  cvt(w_dt,  dtwb, I_DIM * R_DIM);
  cvt(w_out, wob,  H_DIM * I_DIM);

  k_inproj <<<dim3(J_DIM / 256, S_LEN / 64), 256, 0, stream>>>(wib, xb, hs_pre_t, gate_t);
  k_conv   <<<I_DIM / 256, 256, 0, stream>>>(hs_pre_t, mask, conv_w, conv_b, hs_t, hs_tb);
  k_xproj  <<<dim3(1, S_LEN / 64), 320, 0, stream>>>(xwb, hs_tb, ssm);
  k_rms    <<<S_LEN, G_DIM, 0, stream>>>(ssm, dt_ln, b_ln, c_ln, tsb, Bmat, Cmat);
  k_dtproj <<<dim3(I_DIM / 128, S_LEN / 64), 256, 0, stream>>>(dtwb, tsb, b_dt, dt_t);
  k_scan_part   <<<dim3(I_DIM / 256, CH), 256, 0, stream>>>(dt_t, hs_t, Bmat, A_log, Pst, Qst);
  k_scan_combine<<<I_DIM * N_ST / 256, 256, 0, stream>>>(Pst, Qst, S0);
  k_scan_final  <<<dim3(I_DIM / 256, CH), 256, 0, stream>>>(dt_t, hs_t, Bmat, Cmat,
                                                            A_log, Dp, gate_t, S0, sob);
  k_outproj<<<dim3(H_DIM / 256, S_LEN / 64), 256, 0, stream>>>(wob, sob, out);
}
